// GAT_71906342469787
// MI455X (gfx1250) — matrix operand order repaired and checked by reading, not compiled, not run on hardware
//
#include <hip/hip_runtime.h>
#include <hip/hip_bf16.h>
#include <math.h>

typedef __attribute__((ext_vector_type(16))) __bf16    v16bf;
typedef __attribute__((ext_vector_type(8)))  float     v8f;
typedef __attribute__((ext_vector_type(8)))  unsigned  v8u;

#define NEG_SLOPE 0.2f

// ---------------------------------------------------------------------------
// Pack weights W[K,NC] (fp32) into bf16 hi/lo WMMA B-fragments, done ONCE.
// Layout: word index = (((tile*(K/32) + kstep)*32 + lane)*8 + v)
//   lane<16:  column n = tile*16+lane,      K pair = kstep*32 + 2v (+8 if v>=4), +1
//   lane>=16: column n = tile*16+lane-16,   K pair = kstep*32 + 8 + 2v (+8 if v>=4), +1
// Each 32-bit word holds the bf16 pair (low half = even K).
// ---------------------------------------------------------------------------
__global__ void pack_w_k(const float* __restrict__ W, unsigned* __restrict__ whi,
                         unsigned* __restrict__ wlo, int K, int NC) {
  int total = (NC / 16) * (K / 32) * 32 * 8;
  int t = blockIdx.x * blockDim.x + threadIdx.x;
  if (t >= total) return;
  int word = t & 7;
  int lane = (t >> 3) & 31;
  int rest = t >> 8;
  int ks   = rest % (K / 32);
  int tile = rest / (K / 32);
  int n    = tile * 16 + (lane & 15);
  int koff = (lane < 16) ? 0 : 8;
  int k    = ks * 32 + koff + 2 * word + ((word >= 4) ? 8 : 0);
  float x0 = W[(size_t)k * NC + n];
  float x1 = W[(size_t)(k + 1) * NC + n];
  __bf16 h0 = (__bf16)x0, h1 = (__bf16)x1;
  float  r0 = x0 - (float)h0, r1 = x1 - (float)h1;
  __bf16 l0 = (__bf16)r0, l1 = (__bf16)r1;
  whi[t] = ((unsigned)__builtin_bit_cast(unsigned short, h1) << 16) |
           (unsigned)__builtin_bit_cast(unsigned short, h0);
  wlo[t] = ((unsigned)__builtin_bit_cast(unsigned short, l1) << 16) |
           (unsigned)__builtin_bit_cast(unsigned short, l0);
}

// ---------------------------------------------------------------------------
// GEMM: Y[nrows,NC] = X[nrows,K] * W (pre-packed bf16 hi/lo fragments).
// One wave per 16-row tile; ALL column tiles accumulated simultaneously so
// the A fragment is converted once per K-step (3 WMMAs per tile per K-step).
// ---------------------------------------------------------------------------
template<int K, int NC>
__global__ void gemm_bf16split(const float* __restrict__ X,
                               const unsigned* __restrict__ whi,
                               const unsigned* __restrict__ wlo,
                               float* __restrict__ Y, int nrows) {
  const int wave   = (blockIdx.x * blockDim.x + threadIdx.x) >> 5;
  const int lane   = threadIdx.x & 31;
  const int ntiles = nrows >> 4;
  if (wave >= ntiles) return;           // wave-uniform: EXEC stays all-ones
  const int row0 = wave << 4;
  const int m    = lane & 15;
  const int gA   = (lane < 16) ? 0 : 8;
  constexpr int NT = NC / 16;

  v8f acc[NT];
  #pragma unroll
  for (int t = 0; t < NT; ++t) acc[t] = {};

  #pragma unroll
  for (int k0 = 0; k0 < K; k0 += 32) {
    // ---- A fragment (16x32), converted once per K-step ----
    const float* pa = X + (size_t)(row0 + m) * K + k0 + gA;
    float av[16];
    float4 t0 = *(const float4*)(pa + 0);
    float4 t1 = *(const float4*)(pa + 4);
    float4 t2 = *(const float4*)(pa + 16);
    float4 t3 = *(const float4*)(pa + 20);
    av[0]=t0.x;  av[1]=t0.y;  av[2]=t0.z;  av[3]=t0.w;
    av[4]=t1.x;  av[5]=t1.y;  av[6]=t1.z;  av[7]=t1.w;
    av[8]=t2.x;  av[9]=t2.y;  av[10]=t2.z; av[11]=t2.w;
    av[12]=t3.x; av[13]=t3.y; av[14]=t3.z; av[15]=t3.w;
    v16bf ahi, alo;
    #pragma unroll
    for (int i = 0; i < 16; ++i) {
      __bf16 h = (__bf16)av[i];
      ahi[i] = h;
      alo[i] = (__bf16)(av[i] - (float)h);
    }
    const int ks = k0 >> 5;
    #pragma unroll
    for (int t = 0; t < NT; ++t) {
      // ---- B fragments: two coalesced 32B loads of pre-packed data ----
      size_t bbase = (((size_t)t * (K / 32) + ks) * 32 + lane) * 8;
      v16bf bhi = __builtin_bit_cast(v16bf, *(const v8u*)(whi + bbase));
      v16bf blo = __builtin_bit_cast(v16bf, *(const v8u*)(wlo + bbase));
      acc[t] = __builtin_amdgcn_wmma_f32_16x16x32_bf16(false, alo, false, bhi,
                                                       (short)0, acc[t], false, false);
      acc[t] = __builtin_amdgcn_wmma_f32_16x16x32_bf16(false, ahi, false, blo,
                                                       (short)0, acc[t], false, false);
      acc[t] = __builtin_amdgcn_wmma_f32_16x16x32_bf16(false, ahi, false, bhi,
                                                       (short)0, acc[t], false, false);
    }
  }
  // ---- store: C/D VGPR r -> row r (lanes 0-15) / 8+r (lanes 16-31) ----
  const int rowoff = (lane < 16) ? 0 : 8;
  #pragma unroll
  for (int t = 0; t < NT; ++t) {
    float* py = Y + (size_t)(row0 + rowoff) * NC + t * 16 + m;
    #pragma unroll
    for (int r = 0; r < 8; ++r) py[(size_t)r * NC] = acc[t][r];
  }
}

// ---------------------------------------------------------------------------
// Per-(node,head): alpha dot products + seed running max with self-loop logit
// ---------------------------------------------------------------------------
__global__ void alpha_init_k(const float* __restrict__ hbuf,
                             const float* __restrict__ a_src,
                             const float* __restrict__ a_dst,
                             float* __restrict__ as, float* __restrict__ ad,
                             float* __restrict__ nmax,
                             int n, int H, int C, int Fv) {
  int t = blockIdx.x * blockDim.x + threadIdx.x;
  if (t >= n * H) return;
  int i = t / H, h = t - i * H;
  const float* hp = hbuf + (size_t)i * C + h * 16;
  const float* s  = a_src + h * Fv;
  const float* d  = a_dst + h * Fv;
  float s1 = 0.f, s2 = 0.f;
  for (int f = 0; f < Fv; ++f) { float v = hp[f]; s1 += v * s[f]; s2 += v * d[f]; }
  as[t] = s1; ad[t] = s2;
  float e = s1 + s2;
  nmax[t] = (e > 0.f) ? e : NEG_SLOPE * e;
}

// float atomic max via integer punning (pure integer atomics; sign-safe)
__device__ __forceinline__ void atomicMaxF(float* addr, float val) {
  if (val >= 0.f) atomicMax((int*)addr, __float_as_int(val));
  else            atomicMin((unsigned int*)addr, __float_as_uint(val));
}

// Edge pass 1: segment max of leaky-relu logits
__global__ void edge_max_k(const int* __restrict__ src, const int* __restrict__ dst,
                           int E, const float* __restrict__ as,
                           const float* __restrict__ ad,
                           float* __restrict__ nmax, int H) {
  int t = blockIdx.x * blockDim.x + threadIdx.x;
  if (t >= E) return;
  int s = src[t], d = dst[t];
  for (int h = 0; h < H; ++h) {
    float e = as[s * H + h] + ad[d * H + h];
    e = (e > 0.f) ? e : NEG_SLOPE * e;
    atomicMaxF(&nmax[d * H + h], e);
  }
}

// Node-level init of sums/agg with the self-loop contribution
__global__ void self_init_k(const float* __restrict__ as, const float* __restrict__ ad,
                            const float* __restrict__ nmax,
                            const float* __restrict__ hbuf,
                            float* __restrict__ ssum, float* __restrict__ agg,
                            int n, int H, int C) {
  int t = blockIdx.x * blockDim.x + threadIdx.x;
  if (t >= n * H) return;
  int i = t / H, h = t - i * H;
  float e = as[t] + ad[t];
  e = (e > 0.f) ? e : NEG_SLOPE * e;
  float w = __expf(e - nmax[t]);
  ssum[t] = w;
  const float* hp = hbuf + (size_t)i * C + h * 16;
  float* ap = agg + (size_t)i * C + h * 16;
  for (int f = 0; f < 16; ++f) ap[f] = w * hp[f];
}

// Edge pass 2: TPE=min(C,32) threads per edge; lanes cover channels so the
// atomics coalesce. Accumulates unnormalized exp-weighted features + exp-sum.
template<int H, int C>
__global__ void edge_agg_k(const int* __restrict__ src, const int* __restrict__ dst,
                           int E, const float* __restrict__ as,
                           const float* __restrict__ ad,
                           const float* __restrict__ nmax,
                           const float* __restrict__ hbuf,
                           float* __restrict__ ssum, float* __restrict__ agg) {
  constexpr int TPE = (C < 32) ? C : 32;
  long long gid = (long long)blockIdx.x * blockDim.x + threadIdx.x;
  int e    = (int)(gid / TPE);
  int lane = (int)(gid % TPE);
  if (e >= E) return;
  int s = src[e], d = dst[e];
  if (lane < H) {
    float ev = as[s * H + lane] + ad[d * H + lane];
    ev = (ev > 0.f) ? ev : NEG_SLOPE * ev;
    float w = __expf(ev - nmax[d * H + lane]);
    atomicAdd(&ssum[d * H + lane], w);
  }
  #pragma unroll
  for (int c = lane; c < C; c += TPE) {
    int h = c >> 4;
    float ev = as[s * H + h] + ad[d * H + h];
    ev = (ev > 0.f) ? ev : NEG_SLOPE * ev;
    float w = __expf(ev - nmax[d * H + h]);
    atomicAdd(&agg[(size_t)d * C + c], w * hbuf[(size_t)s * C + c]);
  }
}

// Node finalize: normalize, +bias, LayerNorm, ELU. One wave per node.
template<int H, int C>
__global__ void finalize_ln_k(const float* __restrict__ agg,
                              const float* __restrict__ ssum,
                              const float* __restrict__ bias,
                              const float* __restrict__ gamma,
                              const float* __restrict__ beta,
                              float* __restrict__ out, int n) {
  int gid = blockIdx.x * blockDim.x + threadIdx.x;
  int i = gid >> 5, lane = gid & 31;
  if (i >= n) return;
  constexpr int P = C / 32;
  float v[P];
  #pragma unroll
  for (int j = 0; j < P; ++j) {
    int c = lane + 32 * j;
    int h = c >> 4;
    v[j] = agg[(size_t)i * C + c] / (ssum[i * H + h] + 1e-16f) + bias[c];
  }
  float sum = 0.f;
  #pragma unroll
  for (int j = 0; j < P; ++j) sum += v[j];
  #pragma unroll
  for (int o = 16; o > 0; o >>= 1) sum += __shfl_xor(sum, o, 32);
  float mu = sum / C;
  float var = 0.f;
  #pragma unroll
  for (int j = 0; j < P; ++j) { float dv = v[j] - mu; var += dv * dv; }
  #pragma unroll
  for (int o = 16; o > 0; o >>= 1) var += __shfl_xor(var, o, 32);
  var /= C;
  float rstd = rsqrtf(var + 1e-5f);
  #pragma unroll
  for (int j = 0; j < P; ++j) {
    int c = lane + 32 * j;
    float y = (v[j] - mu) * rstd * gamma[c] + beta[c];
    out[(size_t)i * C + c] = (y > 0.f) ? y : (__expf(y) - 1.f);  // ELU
  }
}

// Pad W3 [64,10] -> [64,16] so the WMMA GEMM path stays uniform.
__global__ void pad_w3_k(const float* __restrict__ W3, float* __restrict__ w3p) {
  int t = blockIdx.x * blockDim.x + threadIdx.x;
  if (t >= 64 * 16) return;
  int k = t >> 4, c = t & 15;
  w3p[t] = (c < 10) ? W3[k * 10 + c] : 0.f;
}

// Layer-3 finalize: normalize, +bias, log_softmax over 10 classes.
__global__ void finalize_l3_k(const float* __restrict__ agg,
                              const float* __restrict__ ssum,
                              const float* __restrict__ b3,
                              float* __restrict__ out, int n) {
  int i = blockIdx.x * blockDim.x + threadIdx.x;
  if (i >= n) return;
  float inv = 1.f / (ssum[i] + 1e-16f);
  float v[10];
  float mx = -1e30f;
  for (int c = 0; c < 10; ++c) {
    v[c] = agg[(size_t)i * 16 + c] * inv + b3[c];
    mx = fmaxf(mx, v[c]);
  }
  float se = 0.f;
  for (int c = 0; c < 10; ++c) se += __expf(v[c] - mx);
  float lse = mx + logf(se);
  for (int c = 0; c < 10; ++c) out[(size_t)i * 10 + c] = v[c] - lse;
}

// ---------------------------------------------------------------------------
extern "C" void kernel_launch(void* const* d_in, const int* in_sizes, int n_in,
                              void* d_out, int out_size, void* d_ws, size_t ws_size,
                              hipStream_t stream) {
  const int N = in_sizes[0] / 128;       // 100000
  const int E = in_sizes[1] / 2;         // 1600000

  const float* x   = (const float*)d_in[0];
  const int*   ei  = (const int*)d_in[1];
  const float* W1  = (const float*)d_in[2];
  const float* as1 = (const float*)d_in[3];
  const float* ad1 = (const float*)d_in[4];
  const float* b1  = (const float*)d_in[5];
  const float* g1  = (const float*)d_in[6];
  const float* be1 = (const float*)d_in[7];
  const float* W2  = (const float*)d_in[8];
  const float* as2 = (const float*)d_in[9];
  const float* ad2 = (const float*)d_in[10];
  const float* b2  = (const float*)d_in[11];
  const float* g2  = (const float*)d_in[12];
  const float* be2 = (const float*)d_in[13];
  const float* W3  = (const float*)d_in[14];
  const float* as3 = (const float*)d_in[15];
  const float* ad3 = (const float*)d_in[16];
  const float* b3  = (const float*)d_in[17];
  float* out = (float*)d_out;

  const int* src = ei;
  const int* dst = ei + E;

  float* ws   = (float*)d_ws;
  float* h1   = ws;                      // 128N
  float* agg1 = ws + (size_t)128 * N;    // 128N (finalized in place -> L2 input)
  float* h2   = ws + (size_t)256 * N;    // 64N
  float* agg2 = ws + (size_t)320 * N;    // 64N (finalized in place -> L3 input)
  float* asb  = ws + (size_t)384 * N;    // 8N
  float* adb  = ws + (size_t)392 * N;    // 8N
  float* nmax = ws + (size_t)400 * N;    // 8N
  float* ssum = ws + (size_t)408 * N;    // 8N
  float* w3p  = ws + (size_t)416 * N;    // 1024 floats
  unsigned* whi1 = (unsigned*)(w3p + 1024);   // 8192 words
  unsigned* wlo1 = whi1 + 8192;               // 8192
  unsigned* whi2 = wlo1 + 8192;               // 4096
  unsigned* wlo2 = whi2 + 4096;               // 4096
  unsigned* whi3 = wlo2 + 4096;               // 512
  unsigned* wlo3 = whi3 + 512;                // 512
  float* h3   = h1;                      // reuse dead h1 region
  float* agg3 = h1 + (size_t)16 * N;

  dim3 blk(256);
  auto cdiv = [](long long a, long long b) { return (unsigned)((a + b - 1) / b); };
  unsigned gGemm = cdiv((long long)(N / 16) * 32, 256);
  unsigned gEdge = cdiv(E, 256);
  unsigned gNode = cdiv((long long)N * 32, 256);

  // ---------------- One-time weight prep (pad + bf16 hi/lo fragment pack) ----
  pad_w3_k<<<cdiv(1024, 256), blk, 0, stream>>>(W3, w3p);
  pack_w_k<<<cdiv(8192, 256), blk, 0, stream>>>(W1, whi1, wlo1, 128, 128);
  pack_w_k<<<cdiv(4096, 256), blk, 0, stream>>>(W2, whi2, wlo2, 128, 64);
  pack_w_k<<<cdiv(512, 256),  blk, 0, stream>>>(w3p, whi3, wlo3, 64, 16);

  // ---------------- Layer 1: 128 -> 8 heads x 16 ----------------
  gemm_bf16split<128, 128><<<gGemm, blk, 0, stream>>>(x, whi1, wlo1, h1, N);
  alpha_init_k<<<cdiv((long long)N * 8, 256), blk, 0, stream>>>(h1, as1, ad1, asb, adb, nmax, N, 8, 128, 16);
  edge_max_k<<<gEdge, blk, 0, stream>>>(src, dst, E, asb, adb, nmax, 8);
  self_init_k<<<cdiv((long long)N * 8, 256), blk, 0, stream>>>(asb, adb, nmax, h1, ssum, agg1, N, 8, 128);
  edge_agg_k<8, 128><<<cdiv((long long)E * 32, 256), blk, 0, stream>>>(src, dst, E, asb, adb, nmax, h1, ssum, agg1);
  finalize_ln_k<8, 128><<<gNode, blk, 0, stream>>>(agg1, ssum, b1, g1, be1, agg1, N);

  // ---------------- Layer 2: 128 -> 4 heads x 16 ----------------
  gemm_bf16split<128, 64><<<gGemm, blk, 0, stream>>>(agg1, whi2, wlo2, h2, N);
  alpha_init_k<<<cdiv((long long)N * 4, 256), blk, 0, stream>>>(h2, as2, ad2, asb, adb, nmax, N, 4, 64, 16);
  edge_max_k<<<gEdge, blk, 0, stream>>>(src, dst, E, asb, adb, nmax, 4);
  self_init_k<<<cdiv((long long)N * 4, 256), blk, 0, stream>>>(asb, adb, nmax, h2, ssum, agg2, N, 4, 64);
  edge_agg_k<4, 64><<<cdiv((long long)E * 32, 256), blk, 0, stream>>>(src, dst, E, asb, adb, nmax, h2, ssum, agg2);
  finalize_ln_k<4, 64><<<gNode, blk, 0, stream>>>(agg2, ssum, b2, g2, be2, agg2, N);

  // ---------------- Layer 3: 64 -> 10 (1 head), log_softmax ----------------
  gemm_bf16split<64, 16><<<gGemm, blk, 0, stream>>>(agg2, whi3, wlo3, h3, N);
  alpha_init_k<<<cdiv((long long)N, 256), blk, 0, stream>>>(h3, as3, ad3, asb, adb, nmax, N, 1, 16, 10);
  edge_max_k<<<gEdge, blk, 0, stream>>>(src, dst, E, asb, adb, nmax, 1);
  self_init_k<<<cdiv((long long)N, 256), blk, 0, stream>>>(asb, adb, nmax, h3, ssum, agg3, N, 1, 16);
  edge_agg_k<1, 16><<<cdiv((long long)E * 16, 256), blk, 0, stream>>>(src, dst, E, asb, adb, nmax, h3, ssum, agg3);
  finalize_l3_k<<<cdiv((long long)N, 256), blk, 0, stream>>>(agg3, ssum, b3, out, N);
}
